// SwinTransformerSys_42571715838450
// MI455X (gfx1250) — compile-verified, matrix operand order checked by reading, not run on hardware
//
#include <hip/hip_runtime.h>
#include <hip/hip_fp16.h>

typedef __attribute__((ext_vector_type(16))) _Float16 v16h;
typedef __attribute__((ext_vector_type(8)))  _Float16 v8h;
typedef __attribute__((ext_vector_type(8)))  float    v8f;
typedef __attribute__((ext_vector_type(4)))  int      v4i;
typedef __attribute__((address_space(1))) v4i* gbl_v4i;   // pointer to AS1 v4i
typedef __attribute__((address_space(3))) v4i* lds_v4i;   // pointer to AS3 v4i

#define WMMA_F16(a, b, c) \
  __builtin_amdgcn_wmma_f32_16x16x32_f16(false, (a), false, (b), (short)0, (c), false, false)

#ifndef __has_builtin
#define __has_builtin(x) 0
#endif
#if __has_builtin(__builtin_amdgcn_global_load_async_to_lds_b128)
#define HAVE_ASYNC_LDS 1
#else
#define HAVE_ASYNC_LDS 0
#endif
#if __has_builtin(__builtin_amdgcn_s_wait_asynccnt)
#define HAVE_WAIT_ASYNC 1
#else
#define HAVE_WAIT_ASYNC 0
#endif

// ---- async global->LDS b128 (ASYNCcnt path), with synchronous fallback ------
__device__ __forceinline__ void async_b128(const _Float16* g, _Float16* l) {
#if HAVE_ASYNC_LDS
  v4i* gg = (v4i*)const_cast<_Float16*>(g);
  v4i* ll = (v4i*)l;
  __builtin_amdgcn_global_load_async_to_lds_b128((gbl_v4i)gg, (lds_v4i)ll, 0, 0);
#else
  *(uint4*)l = *(const uint4*)g;
#endif
}
__device__ __forceinline__ void wait_async_le4() {
#if HAVE_WAIT_ASYNC
  __builtin_amdgcn_s_wait_asynccnt(4);
#elif HAVE_ASYNC_LDS
  asm volatile("s_wait_asynccnt 0x4" ::: "memory");
#endif
}
__device__ __forceinline__ void wait_async_le0() {
#if HAVE_WAIT_ASYNC
  __builtin_amdgcn_s_wait_asynccnt(0);
#elif HAVE_ASYNC_LDS
  asm volatile("s_wait_asynccnt 0x0" ::: "memory");
#endif
}

// ---- CDNA5 16-bit fragment loader -------------------------------------------
// A 16x32 (MxK): lane 0-15 -> M=lane, lanes 16-31 -> M=lane-16.
// Per lane: v16h = { K=kb..kb+7 , K=kb+16..kb+23 }, kb = (lane>=16)*8.
// B 32x16 (KxN) stored transposed [N][K]: same chunking, row = column index.
__device__ __forceinline__ v16h load_frag(const _Float16* base, int row, int ldk, int kb) {
  const _Float16* p = base + row * ldk + kb;
  v8h lo = *(const v8h*)(p);
  v8h hi = *(const v8h*)(p + 16);
  return __builtin_shufflevector(lo, hi, 0,1,2,3,4,5,6,7,8,9,10,11,12,13,14,15);
}

// ---- weight fp32 -> f16, transposed to [N][K], LDS-tiled (coalesced both ways)
__global__ __launch_bounds__(256) void wconv_kernel(const float* __restrict__ src,
                                                    _Float16* __restrict__ dst,
                                                    int K, int N) {
  __shared__ float tile[32][33];
  int k0 = blockIdx.y * 32, n0 = blockIdx.x * 32;
  int t = threadIdx.x;
#pragma unroll
  for (int idx = t; idx < 1024; idx += 256) {
    int r = idx >> 5, c = idx & 31;                  // r over K, c over N (coalesced)
    tile[r][c] = src[(size_t)(k0 + r) * N + (n0 + c)];
  }
  __syncthreads();
#pragma unroll
  for (int idx = t; idx < 1024; idx += 256) {
    int r = idx >> 5, c = idx & 31;                  // r over N, c over K (coalesced)
    dst[(size_t)(n0 + r) * K + (k0 + c)] = (_Float16)tile[c][r];
  }
}

// ---- LayerNorm (wave per row); shiftMode=1: fused roll(-4,-4)+window gather --
__global__ __launch_bounds__(256) void ln_kernel(const float* __restrict__ x,
                                                 const float* __restrict__ gam,
                                                 const float* __restrict__ bet,
                                                 _Float16* __restrict__ out,
                                                 int shiftMode) {
  int lane = threadIdx.x & 31, wv = threadIdx.x >> 5;
  int r = blockIdx.x * 8 + wv;                       // destination (windowed) row
  size_t srow;
  if (shiftMode) {
    int b = r >> 12, rem = r & 4095, win = rem >> 6, n = rem & 63;
    int wh = win >> 3, ww = win & 7, i = n >> 3, j = n & 7;
    int hh = (wh * 8 + i + 4) & 63;                  // roll(-SHIFT): src = dst+4 mod 64
    int wc = (ww * 8 + j + 4) & 63;
    srow = ((size_t)b << 12) + (size_t)hh * 64 + wc;
  } else {
    srow = (size_t)r;
  }
  const float* px = x + srow * 512;
  float v[16], sum = 0.f, sq = 0.f;
#pragma unroll
  for (int k = 0; k < 16; ++k) { float t = px[lane + 32 * k]; v[k] = t; sum += t; sq += t * t; }
#pragma unroll
  for (int off = 16; off; off >>= 1) { sum += __shfl_xor(sum, off, 32); sq += __shfl_xor(sq, off, 32); }
  float mean = sum * (1.f / 512.f);
  float rstd = rsqrtf(sq * (1.f / 512.f) - mean * mean + 1e-5f);
#pragma unroll
  for (int k = 0; k < 16; ++k) {
    int c = lane + 32 * k;
    out[(size_t)r * 512 + c] = (_Float16)((v[k] - mean) * rstd * gam[c] + bet[c]);
  }
}

// ---- generic WMMA GEMM: out[M,N] = A[M,K] @ Bt[N,K]^T + bias -----------------
// Double-buffered LDS, async global->LDS tile copies (ASYNCcnt pipelined).
// mode 0: f16 store            mode 2: GELU -> f16 store
// mode 1: window-reverse + unshift scatter + residual -> f32 store
// mode 3: accumulate into f32 out
__global__ __launch_bounds__(256) void gemm_wmma(const _Float16* __restrict__ A,
                                                 const _Float16* __restrict__ Bt,
                                                 const float* __restrict__ bias,
                                                 const float* __restrict__ resid,
                                                 _Float16* __restrict__ outH,
                                                 float* __restrict__ outF,
                                                 int M, int N, int K, int mode) {
  __shared__ __align__(16) _Float16 sA[2][128 * 32];
  __shared__ __align__(16) _Float16 sB[2][128 * 32];
  int t = threadIdx.x, lane = t & 31, wv = t >> 5;
  int wrow = wv >> 2, wcol = wv & 3;                 // 2x4 wave grid, wave tile 64x32
  int mBase = blockIdx.y * 128, nBase = blockIdx.x * 128;
  int l15 = lane & 15, kb = (lane >> 4) * 8;
  int cr = t >> 2, cs = (t & 3) * 8;                 // per-thread tile chunk: rows cr, cr+64
  v8f acc[4][2] = {};

  auto load_tile = [&](int k0, int buf) {
    async_b128(&A[(size_t)(mBase + cr) * K + k0 + cs],        &sA[buf][cr * 32 + cs]);
    async_b128(&A[(size_t)(mBase + cr + 64) * K + k0 + cs],   &sA[buf][(cr + 64) * 32 + cs]);
    async_b128(&Bt[(size_t)(nBase + cr) * K + k0 + cs],       &sB[buf][cr * 32 + cs]);
    async_b128(&Bt[(size_t)(nBase + cr + 64) * K + k0 + cs],  &sB[buf][(cr + 64) * 32 + cs]);
#if !HAVE_ASYNC_LDS
    if (k0 + 32 < K) {                               // -> global_prefetch_b8
      __builtin_prefetch(&A[(size_t)(mBase + cr) * K + k0 + 32], 0, 1);
      __builtin_prefetch(&Bt[(size_t)(nBase + cr) * K + k0 + 32], 0, 1);
    }
#endif
  };

  int nk = K >> 5;
  load_tile(0, 0);
  for (int i = 0; i < nk; ++i) {
    if (i + 1 < nk) {                                // pipeline next tile into other buffer
      load_tile((i + 1) << 5, (i + 1) & 1);
      wait_async_le4();                              // tile i resident (in-order retire)
    } else {
      wait_async_le0();
    }
    __syncthreads();
    const _Float16* cA = sA[i & 1];
    const _Float16* cB = sB[i & 1];
    v16h bf0 = load_frag(cB, wcol * 32 + l15, 32, kb);
    v16h bf1 = load_frag(cB, wcol * 32 + 16 + l15, 32, kb);
    v16h af0 = load_frag(cA, wrow * 64 + l15, 32, kb);
    v16h af1 = load_frag(cA, wrow * 64 + 16 + l15, 32, kb);
    v16h af2 = load_frag(cA, wrow * 64 + 32 + l15, 32, kb);
    v16h af3 = load_frag(cA, wrow * 64 + 48 + l15, 32, kb);
    acc[0][0] = WMMA_F16(af0, bf0, acc[0][0]);
    acc[0][1] = WMMA_F16(af0, bf1, acc[0][1]);
    acc[1][0] = WMMA_F16(af1, bf0, acc[1][0]);
    acc[1][1] = WMMA_F16(af1, bf1, acc[1][1]);
    acc[2][0] = WMMA_F16(af2, bf0, acc[2][0]);
    acc[2][1] = WMMA_F16(af2, bf1, acc[2][1]);
    acc[3][0] = WMMA_F16(af3, bf0, acc[3][0]);
    acc[3][1] = WMMA_F16(af3, bf1, acc[3][1]);
    __syncthreads();                                 // all waves done reading buf (i&1)
  }

  int hi8 = (lane >= 16) ? 8 : 0;
#pragma unroll
  for (int mt = 0; mt < 4; ++mt)
#pragma unroll
    for (int nt = 0; nt < 2; ++nt)
#pragma unroll
      for (int r = 0; r < 8; ++r) {
        int grow = mBase + wrow * 64 + mt * 16 + r + hi8;
        int gcol = nBase + wcol * 32 + nt * 16 + l15;
        float v = acc[mt][nt][r] + bias[gcol];
        if (mode == 0) {
          outH[(size_t)grow * N + gcol] = (_Float16)v;
        } else if (mode == 2) {
          float g = 0.5f * v * (1.f + erff(v * 0.70710678118f));   // exact GELU
          outH[(size_t)grow * N + gcol] = (_Float16)g;
        } else if (mode == 1) {
          int b = grow >> 12, rem = grow & 4095, win = rem >> 6, n = rem & 63;
          int wh = win >> 3, ww = win & 7, i = n >> 3, j = n & 7;
          int hh = (wh * 8 + i + 4) & 63;           // roll(+SHIFT)
          int wc = (ww * 8 + j + 4) & 63;
          size_t tok = ((size_t)b << 12) + (size_t)hh * 64 + wc;
          outF[tok * 512 + gcol] = resid[tok * 512 + gcol] + v;
        } else {
          outF[(size_t)grow * N + gcol] += v;
        }
      }
}

// ---- windowed attention: one block per (window, head) ------------------------
__global__ __launch_bounds__(128) void attn_kernel(const _Float16* __restrict__ qkv,
                                                   const float* __restrict__ rpb,
                                                   _Float16* __restrict__ outA) {
  __shared__ __align__(16) _Float16 sQ[64 * 32];
  __shared__ __align__(16) _Float16 sK[64 * 32];
  __shared__ __align__(16) _Float16 sVT[32 * 72];   // v transposed [d][m]
  __shared__ __align__(16) float    sS[64 * 72];    // scores
  __shared__ __align__(16) _Float16 sP[64 * 72];    // softmax probs (A layout)
  int t = threadIdx.x, lane = t & 31, wv = t >> 5;
  int win = blockIdx.x, head = blockIdx.y;
  int w64 = win & 63, wh = w64 >> 3, ww = w64 & 7;
  size_t rowBase = (size_t)win * 64;
  int qoff = head * 32, koff = 512 + head * 32, voff = 1024 + head * 32;

  {  // stage q,k row-major; v transposed
    int row = t >> 1, seg = t & 1;
    const _Float16* src = qkv + (rowBase + row) * 1536;
    *(uint4*)&sQ[row * 32 + seg * 16]     = *(const uint4*)&src[qoff + seg * 16];
    *(uint4*)&sQ[row * 32 + seg * 16 + 8] = *(const uint4*)&src[qoff + seg * 16 + 8];
    *(uint4*)&sK[row * 32 + seg * 16]     = *(const uint4*)&src[koff + seg * 16];
    *(uint4*)&sK[row * 32 + seg * 16 + 8] = *(const uint4*)&src[koff + seg * 16 + 8];
#pragma unroll
    for (int e = 0; e < 16; ++e) { int d = seg * 16 + e; sVT[d * 72 + row] = src[voff + d]; }
  }
  __syncthreads();

  int kb = (lane >> 4) * 8, hi8 = (lane >= 16) ? 8 : 0, l15 = lane & 15;

  // pass 1: scores = q @ k^T  (K = 32 = head dim, single WMMA step)
  v8f sc[4] = {};
  v16h aq = load_frag(sQ, wv * 16 + l15, 32, kb);
#pragma unroll
  for (int nt = 0; nt < 4; ++nt) {
    v16h bk = load_frag(sK, nt * 16 + l15, 32, kb);
    sc[nt] = WMMA_F16(aq, bk, sc[nt]);
  }
#pragma unroll
  for (int nt = 0; nt < 4; ++nt)
#pragma unroll
    for (int r = 0; r < 8; ++r) {
      int qrow = wv * 16 + r + hi8, kcol = nt * 16 + l15;
      int i1 = qrow >> 3, j1 = qrow & 7, i2 = kcol >> 3, j2 = kcol & 7;
      float bias = rpb[((i1 - i2 + 7) * 15 + (j1 - j2 + 7)) * 16 + head];
      int g1 = ((wh < 7) ? 0 : ((i1 < 4) ? 1 : 2)) * 3 + ((ww < 7) ? 0 : ((j1 < 4) ? 1 : 2));
      int g2 = ((wh < 7) ? 0 : ((i2 < 4) ? 1 : 2)) * 3 + ((ww < 7) ? 0 : ((j2 < 4) ? 1 : 2));
      float msk = (g1 != g2) ? -100.f : 0.f;
      sS[qrow * 72 + kcol] = sc[nt][r] * 0.17677669529663687f + bias + msk;  // *HD^-0.5
    }
  __syncthreads();

  if (t < 64) {  // softmax per query row
    float mx = -1e30f;
    for (int m = 0; m < 64; ++m) mx = fmaxf(mx, sS[t * 72 + m]);
    float sum = 0.f;
    for (int m = 0; m < 64; ++m) { float e = __expf(sS[t * 72 + m] - mx); sum += e; sS[t * 72 + m] = e; }
    float inv = 1.f / sum;
    for (int m = 0; m < 64; ++m) sP[t * 72 + m] = (_Float16)(sS[t * 72 + m] * inv);
  }
  __syncthreads();

  // pass 2: out = attn @ v  (K = 64 -> two WMMA steps)
  v8f oc[2] = {};
#pragma unroll
  for (int s2 = 0; s2 < 2; ++s2) {
    v16h ap = load_frag(sP, wv * 16 + l15, 72, kb + 32 * s2);
#pragma unroll
    for (int nt = 0; nt < 2; ++nt) {
      v16h bv = load_frag(sVT, nt * 16 + l15, 72, kb + 32 * s2);
      oc[nt] = WMMA_F16(ap, bv, oc[nt]);
    }
  }
#pragma unroll
  for (int nt = 0; nt < 2; ++nt)
#pragma unroll
    for (int r = 0; r < 8; ++r) {
      int qrow = wv * 16 + r + hi8, d = nt * 16 + l15;
      outA[(rowBase + qrow) * 512 + head * 32 + d] = (_Float16)(oc[nt][r]);
    }
}

extern "C" void kernel_launch(void* const* d_in, const int* in_sizes, int n_in,
                              void* d_out, int out_size, void* d_ws, size_t ws_size,
                              hipStream_t stream) {
  (void)in_sizes; (void)n_in; (void)out_size; (void)ws_size;
  const float* x     = (const float*)d_in[0];
  const float* n1w   = (const float*)d_in[1];
  const float* n1b   = (const float*)d_in[2];
  const float* qkvw  = (const float*)d_in[3];
  const float* qkvb  = (const float*)d_in[4];
  const float* projw = (const float*)d_in[5];
  const float* projb = (const float*)d_in[6];
  const float* rpb   = (const float*)d_in[7];
  const float* n2w   = (const float*)d_in[8];
  const float* n2b   = (const float*)d_in[9];
  const float* fc1w  = (const float*)d_in[10];
  const float* fc1b  = (const float*)d_in[11];
  const float* fc2w  = (const float*)d_in[12];
  const float* fc2b  = (const float*)d_in[13];
  float* out = (float*)d_out;

  char* ws = (char*)d_ws;
  _Float16* bufA = (_Float16*)ws;                                   // 32768x512  f16 (33.5 MB)
  _Float16* bufB = (_Float16*)(ws + (size_t)33554432);              // 32768x2048 f16 (134 MB)
  _Float16* wq = (_Float16*)(ws + (size_t)33554432 + 134217728);    // f16 weights, [N][K]
  _Float16* wp = wq + 1536 * 512;
  _Float16* w1 = wp + 512 * 512;
  _Float16* w2 = w1 + 2048 * 512;

  // 0) weights -> f16 transposed [N][K] (tiled transpose, coalesced)
  wconv_kernel<<<dim3(1536 / 32, 512 / 32), 256, 0, stream>>>(qkvw, wq, 512, 1536);
  wconv_kernel<<<dim3(512 / 32, 512 / 32), 256, 0, stream>>>(projw, wp, 512, 512);
  wconv_kernel<<<dim3(2048 / 32, 512 / 32), 256, 0, stream>>>(fc1w, w1, 512, 2048);
  wconv_kernel<<<dim3(512 / 32, 2048 / 32), 256, 0, stream>>>(fc2w, w2, 2048, 512);

  // 1) LN1 + cyclic shift + window partition -> bufA
  ln_kernel<<<4096, 256, 0, stream>>>(x, n1w, n1b, bufA, 1);
  // 2) qkv GEMM -> bufB
  gemm_wmma<<<dim3(1536 / 128, 32768 / 128), 256, 0, stream>>>(
      bufA, wq, qkvb, nullptr, bufB, nullptr, 32768, 1536, 512, 0);
  // 3) windowed attention -> bufA (bufA input fully consumed by step 2)
  attn_kernel<<<dim3(512, 16), 128, 0, stream>>>(bufB, rpb, bufA);
  // 4) proj GEMM + window-reverse + unshift + residual -> d_out (x1)
  gemm_wmma<<<dim3(512 / 128, 32768 / 128), 256, 0, stream>>>(
      bufA, wp, projb, x, nullptr, out, 32768, 512, 512, 1);
  // 5) LN2 -> bufA
  ln_kernel<<<4096, 256, 0, stream>>>(out, n2w, n2b, bufA, 0);
  // 6) fc1 GEMM + GELU -> bufB
  gemm_wmma<<<dim3(2048 / 128, 32768 / 128), 256, 0, stream>>>(
      bufA, w1, fc1b, nullptr, bufB, nullptr, 32768, 2048, 512, 2);
  // 7) fc2 GEMM, accumulate into d_out
  gemm_wmma<<<dim3(512 / 128, 32768 / 128), 256, 0, stream>>>(
      bufB, w2, fc2b, nullptr, nullptr, out, 32768, 512, 2048, 3);
}
